// SparseHierarchicalGNNAttention_7559142441641
// MI455X (gfx1250) — matrix operand order repaired and checked by reading, not compiled, not run on hardware
//
#include <hip/hip_runtime.h>
#include <math.h>

// ---------------------------------------------------------------------------
// CDNA5 / gfx1250 WMMA helpers (wave32). D = A(16x32 bf16) x B(32x16 bf16) + C
// All operands live in "fragment-major" layouts: frag[lane][elems] contiguous,
// so every fragment load/store is wide (b128) with zero packing VALU.
// ---------------------------------------------------------------------------
typedef __attribute__((ext_vector_type(16))) __bf16 v16bf;
typedef __attribute__((ext_vector_type(8)))  __bf16 v8bf;
typedef __attribute__((ext_vector_type(8)))  float  v8f;

__device__ __forceinline__ v8f wmma_bf16(v16bf a, v16bf b, v8f c) {
  // 8 args: (neg_a, A, neg_b, B, c_mod, C, reuse_a, reuse_b)
  return __builtin_amdgcn_wmma_f32_16x16x32_bf16(false, a, false, b,
                                                 (short)0, c, false, false);
}

// --- WMMA operand element maps (CDNA5 ISA §7.12.2) ---------------
// A 16x32 bf16: lane = m + 16*half, half = (kk>>3)&1 ; elem = (kk&7) + ((kk&16)?8:0)
__device__ __forceinline__ int a_swz_slot(int m, int kk) {
  int lane = m + (((kk >> 3) & 1) << 4);
  int i = (kk & 7) + ((kk & 16) ? 8 : 0);
  return lane * 16 + i;
}
// B 32x16 bf16: lane = n + 16*((kk>>3)&1) ; elem = (kk&7) + ((kk&16)?8:0)
__device__ __forceinline__ size_t b_swz_index(int k, int n, int KT) {
  int kt = k >> 5, kk = k & 31;
  int lane = (n & 15) + (((kk >> 3) & 1) << 4);
  int nt = n >> 4;
  return ((size_t)(nt * KT + kt) * 32 + lane) * 16 + ((kk & 7) + ((kk & 16) ? 8 : 0));
}
// C/D 16x16 f32 fragment-major store: tile = [32 lanes][8 f32] (1 KB/tile)
__device__ __forceinline__ void store_c_frag(float* Cf, int tile, int lane, v8f c) {
  ((v8f*)Cf)[tile * 32 + lane] = c;   // 2x ds_store_b128
}
// read element (row r, global col c) from C-fragment-major buffer
__device__ __forceinline__ int c_frag_idx(int r, int c) {
  // tile = c>>4 ; lane_c = (c&15) + 16*(r>=8) ; elem = r&7
  return (c >> 4) * 256 + (((c & 15) + ((r >> 3) << 4)) << 3) + (r & 7);
}
// C/D row-major store (only used for the tiny 16x16 emb tile)
__device__ __forceinline__ void store_c(float* H, int ldn, int n0, int lane, v8f c) {
  int n  = lane & 15;
  int mb = (lane >> 4) ? 8 : 0;
#pragma unroll
  for (int i = 0; i < 8; ++i) H[(mb + i) * ldn + n0 + n] = c[i];
}

__device__ __forceinline__ float wave_sum(float v) {
#pragma unroll
  for (int m = 16; m > 0; m >>= 1) v += __shfl_xor(v, m, 32);
  return v;
}

// Order-preserving float<->uint encoding for atomic segment min/max.
__device__ __forceinline__ unsigned enc_f(float f) {
  unsigned u = __float_as_uint(f);
  return (u & 0x80000000u) ? ~u : (u | 0x80000000u);
}
__device__ __forceinline__ float dec_f(unsigned u) {
  return __uint_as_float((u & 0x80000000u) ? (u ^ 0x80000000u) : ~u);
}

#define LN_EPS 1e-5f
#define ATT_EPS 1e-4f

// ---------------------------------------------------------------------------
// init: zero accumulators, flood min/max encodings
// ---------------------------------------------------------------------------
__global__ void init_kernel(float* sums, float* cnt, float* asum, float* out_sup,
                            unsigned* amaxe, unsigned* amine,
                            unsigned* smaxe, unsigned* smine, int Cn, int Nn) {
  int g = blockIdx.x * blockDim.x + threadIdx.x;
  int stride = gridDim.x * blockDim.x;
  for (int i = g; i < Cn * 16; i += stride) sums[i] = 0.f;
  for (int i = g; i < Cn; i += stride) { cnt[i] = 0.f; smaxe[i] = 0u; smine[i] = 0xFFFFFFFFu; }
  for (int i = g; i < Nn; i += stride) { asum[i] = 0.f; amaxe[i] = 0u; amine[i] = 0xFFFFFFFFu; }
  for (int i = g; i < Cn * 128; i += stride) out_sup[i] = 0.f;
}

// fp32 -> bf16 weight conversion + swizzle into WMMA-B fragment-major layout.
// src: row-major [K][Nw] with Nw = 1<<nsh.
__global__ void swz_kernel(const float* __restrict__ s, __bf16* __restrict__ d,
                           int K, int nsh) {
  int g = blockIdx.x * blockDim.x + threadIdx.x;
  int stride = gridDim.x * blockDim.x;
  int KT = K >> 5;
  int total = K << nsh;
  int nmask = (1 << nsh) - 1;
  for (int idx = g; idx < total; idx += stride) {
    int k = idx >> nsh, n = idx & nmask;
    d[b_swz_index(k, n, KT)] = (__bf16)s[idx];
  }
}

// ---------------------------------------------------------------------------
// Shared LN helper: H is C-fragment-major [ntiles][32][8]; lane owns the 16
// contiguous columns c = lane*16+j of a 512-wide row. Writes relu(LN(x)) as
// bf16 into A-fragment-major Hbf via two packed 16-byte stores per row.
// ---------------------------------------------------------------------------
__device__ __forceinline__ void ln_relu_to_frags_512(
    const float* H, __bf16* Hbf, int lane,
    const float* __restrict__ bias, const float* __restrict__ gamma,
    const float* __restrict__ beta) {
  float br[16], gr[16], ber[16];
#pragma unroll
  for (int j = 0; j < 16; ++j) {
    int c = lane * 16 + j;              // contiguous -> b128 loads
    br[j] = bias[c]; gr[j] = gamma[c]; ber[j] = beta[c];
  }
  int kt = lane >> 1;                   // c>>5 for c = lane*16+j, j<16
  int ibase = (lane & 1) * 8;           // elem offset within A fragment
  for (int r = 0; r < 16; ++r) {
    float vv[16];
    float s = 0.f, s2 = 0.f;
#pragma unroll
    for (int j = 0; j < 16; ++j) {
      // element (r, lane*16+j): tile = lane, col-in-tile = j
      float v = H[lane * 256 + ((j + ((r >> 3) << 4)) << 3) + (r & 7)] + br[j];
      vv[j] = v; s += v; s2 += v * v;
    }
    s = wave_sum(s); s2 = wave_sum(s2);
    float mean = s * (1.f / 512.f);
    float var  = s2 * (1.f / 512.f) - mean * mean;
    float inv  = rsqrtf(var + LN_EPS);
    v8bf p0, p1;
#pragma unroll
    for (int j = 0; j < 8; ++j) {
      float v = (vv[j] - mean) * inv * gr[j] + ber[j];
      p0[j] = (__bf16)fmaxf(v, 0.f);
    }
#pragma unroll
    for (int j = 8; j < 16; ++j) {
      float v = (vv[j] - mean) * inv * gr[j] + ber[j];
      p1[j - 8] = (__bf16)fmaxf(v, 0.f);
    }
    // A-fragment slots: rows r (half 0) and r+16 (half 1), elems ibase..ibase+7
    *(v8bf*)(Hbf + kt * 512 + r * 16 + ibase)        = p0;
    *(v8bf*)(Hbf + kt * 512 + (r + 16) * 16 + ibase) = p1;
  }
}

// ---------------------------------------------------------------------------
// Node clustering MLP: emb = l2norm(relu(LN(nodes@cW1+cb1))@cW2 + cb2)
// One wave per 16 nodes. LATENT=128, HIDDEN=512, EMB=16.
// ---------------------------------------------------------------------------
__global__ __launch_bounds__(32)
void node_mlp_kernel(const float* __restrict__ nodes,
                     const __bf16* __restrict__ cW1z, const float* __restrict__ cb1,
                     const float* __restrict__ cg1, const float* __restrict__ cbe1,
                     const __bf16* __restrict__ cW2z, const float* __restrict__ cb2,
                     float* __restrict__ emb, int Nn) {
  __shared__ __align__(32) char smem[49152];
  float*  H   = (float*)smem;               // 32 C-frags (32 KB)
  __bf16* Af  = (__bf16*)(smem + 32768);    // 4 A-frags  (4 KB)
  __bf16* Hbf = (__bf16*)(smem + 32768);    // 16 A-frags (16 KB, reuses Af)
  int lane = threadIdx.x;
  int row0 = blockIdx.x * 16;

  // stage A tile in fragment order (fp32 -> bf16)
  for (int idx = lane; idx < 16 * 128; idx += 32) {
    int r = idx >> 7, k = idx & 127;
    int gr = row0 + r;
    float v = (gr < Nn) ? nodes[(size_t)gr * 128 + k] : 0.f;
    Af[(k >> 5) * 512 + a_swz_slot(r, k & 31)] = (__bf16)v;
  }
  __syncthreads();

  const v16bf* AfV = (const v16bf*)Af;
  const v16bf* W1V = (const v16bf*)cW1z;  // KT=4
  // layer 1: H = A[16x128] @ cW1[128x512]  (32 n-tiles x 4 k-steps)
  for (int nt = 0; nt < 32; ++nt) {
    v8f acc = {};
#pragma unroll
    for (int ks = 0; ks < 4; ++ks) {
      v16bf a = AfV[ks * 32 + lane];
      v16bf b = W1V[(size_t)(nt * 4 + ks) * 32 + lane];
      acc = wmma_bf16(a, b, acc);
    }
    store_c_frag(H, nt, lane, acc);
  }
  __syncthreads();

  // bias + LayerNorm + ReLU -> bf16 A-fragments (Hbf overwrites Af; Af is dead)
  ln_relu_to_frags_512(H, Hbf, lane, cb1, cg1, cbe1);
  __syncthreads();

  // layer 2: E[16x16] = Hb[16x512] @ cW2[512x16]  (16 k-steps, KT=16, nt=0)
  const v16bf* HbV = (const v16bf*)Hbf;
  const v16bf* W2V = (const v16bf*)cW2z;
  v8f acc2 = {};
#pragma unroll
  for (int ks = 0; ks < 16; ++ks) {
    v16bf a = HbV[ks * 32 + lane];
    v16bf b = W2V[(size_t)ks * 32 + lane];
    acc2 = wmma_bf16(a, b, acc2);
  }
  float* E = (float*)smem;  // H region is dead now
  __syncthreads();
  store_c(E, 16, 0, lane, acc2);   // row-major: contiguous row reads below
  __syncthreads();

  // bias + row L2-normalize, write emb
  if (lane < 16) {
    float v[16], ss = 0.f;
#pragma unroll
    for (int d = 0; d < 16; ++d) { v[d] = E[lane * 16 + d] + cb2[d]; ss += v[d] * v[d]; }
    float nrm = fmaxf(sqrtf(ss), 1e-12f);
    int gr = row0 + lane;
    if (gr < Nn)
#pragma unroll
      for (int d = 0; d < 16; ++d) emb[(size_t)gr * 16 + d] = v[d] / nrm;
  }
}

// ---------------------------------------------------------------------------
// scatter-mean of emb per cluster -> means
// ---------------------------------------------------------------------------
__global__ void scatter_sum_kernel(const float* __restrict__ emb,
                                   const int* __restrict__ clusters,
                                   float* sums, float* cnt, int Nn, int Cn) {
  int idx = blockIdx.x * blockDim.x + threadIdx.x;
  if (idx >= Nn * 16) return;
  int i = idx >> 4, d = idx & 15;
  int c = clusters[i];
  if ((unsigned)c >= (unsigned)Cn) return;
  atomicAdd(&sums[c * 16 + d], emb[idx]);
  if (d == 0) atomicAdd(&cnt[c], 1.f);
}

__global__ void means_kernel(const float* sums, const float* cnt, float* means, int Cn) {
  int c = blockIdx.x * blockDim.x + threadIdx.x;
  if (c >= Cn) return;
  float k = fmaxf(cnt[c], 1.f);
  float v[16], ss = 0.f;
#pragma unroll
  for (int d = 0; d < 16; ++d) { v[d] = sums[c * 16 + d] / k; ss += v[d] * v[d]; }
  float nrm = fmaxf(sqrtf(ss), 1e-12f);
#pragma unroll
  for (int d = 0; d < 16; ++d) means[c * 16 + d] = v[d] / nrm;
}

// ---------------------------------------------------------------------------
// bipartite attention: 3 passes over EB edges (seg min/max, exp+sum, normalize)
// ---------------------------------------------------------------------------
__global__ void bip_dot_kernel(const float* __restrict__ emb, const float* __restrict__ means,
                               const int* __restrict__ src, const int* __restrict__ dst,
                               float* att, unsigned* amaxe, unsigned* amine, int EBn) {
  int e = blockIdx.x * blockDim.x + threadIdx.x;
  if (e >= EBn) return;
  int s = src[e], d = dst[e];
  const float4* a = (const float4*)(emb + (size_t)s * 16);
  const float4* b = (const float4*)(means + (size_t)d * 16);
  float dot = 0.f;
#pragma unroll
  for (int i = 0; i < 4; ++i) {
    float4 x = a[i], y = b[i];
    dot += x.x * y.x + x.y * y.y + x.z * y.z + x.w * y.w;
  }
  att[e] = dot;
  unsigned enc = enc_f(dot);
  atomicMax(&amaxe[s], enc);
  atomicMin(&amine[s], enc);
}

__global__ void bip_exp_kernel(float* att, const unsigned* amaxe, const unsigned* amine,
                               const int* __restrict__ src, float* asum, int EBn) {
  int e = blockIdx.x * blockDim.x + threadIdx.x;
  if (e >= EBn) return;
  int s = src[e];
  float a = att[e];
  float mx = dec_f(amaxe[s]), mn = dec_f(amine[s]);
  float v = expf(2.f * (a - mn) / (ATT_EPS + (mx - mn)));
  att[e] = v;
  atomicAdd(&asum[s], v);
}

__global__ void bip_norm_kernel(float* att, const float* asum,
                                const int* __restrict__ src, float* out_bip, int EBn) {
  int e = blockIdx.x * blockDim.x + threadIdx.x;
  if (e >= EBn) return;
  float w = att[e] / (ATT_EPS + asum[src[e]]);
  att[e] = w;          // reused by supernode scatter
  out_bip[e] = w;
}

// supernodes[dst] += nodes[src] * w  — one wave per edge, 4 floats/lane
__global__ __launch_bounds__(256)
void scatter_sup_kernel(const float* __restrict__ nodes, const float* __restrict__ w,
                        const int* __restrict__ src, const int* __restrict__ dst,
                        float* sup, int EBn) {
  int wid  = (blockIdx.x * blockDim.x + threadIdx.x) >> 5;
  int lane = threadIdx.x & 31;
  if (wid >= EBn) return;
  int s = src[wid], d = dst[wid];
  float we = w[wid];
  float4 v = ((const float4*)(nodes + (size_t)s * 128))[lane];
  float* out = sup + (size_t)d * 128 + lane * 4;
  atomicAdd(out + 0, v.x * we);
  atomicAdd(out + 1, v.y * we);
  atomicAdd(out + 2, v.z * we);
  atomicAdd(out + 3, v.w * we);
}

// ---------------------------------------------------------------------------
// supergraph attention (grouped by sup_dst), tanh normalization
// ---------------------------------------------------------------------------
__global__ void sup_dot_kernel(const float* __restrict__ means,
                               const int* __restrict__ ssrc, const int* __restrict__ sdst,
                               float* satt, unsigned* smaxe, unsigned* smine, int ESn) {
  int e = blockIdx.x * blockDim.x + threadIdx.x;
  if (e >= ESn) return;
  int s = ssrc[e], d = sdst[e];
  const float4* a = (const float4*)(means + (size_t)s * 16);
  const float4* b = (const float4*)(means + (size_t)d * 16);
  float dot = 0.f;
#pragma unroll
  for (int i = 0; i < 4; ++i) {
    float4 x = a[i], y = b[i];
    dot += x.x * y.x + x.y * y.y + x.z * y.z + x.w * y.w;
  }
  satt[e] = dot;
  unsigned enc = enc_f(dot);
  atomicMax(&smaxe[d], enc);
  atomicMin(&smine[d], enc);
}

__global__ void sup_tanh_kernel(const float* satt, const unsigned* smaxe, const unsigned* smine,
                                const int* __restrict__ sdst, float* out_satt, int ESn) {
  int e = blockIdx.x * blockDim.x + threadIdx.x;
  if (e >= ESn) return;
  int d = sdst[e];
  float mx = dec_f(smaxe[d]), mn = dec_f(smine[d]);
  out_satt[e] = tanhf(2.f * (satt[e] - mn) / (ATT_EPS + (mx - mn)));
}

// ---------------------------------------------------------------------------
// Superedge encoder: relu(LN(relu(LN(concat(sup[s],sup[d])@eW1+eb1))@eW2+eb2))
// One wave per 16 edges. 2*LATENT=256, HIDDEN=512, out=128.
// ---------------------------------------------------------------------------
__global__ __launch_bounds__(32)
void superedge_mlp_kernel(const float* __restrict__ sup,
                          const int* __restrict__ ssrc, const int* __restrict__ sdst,
                          const __bf16* __restrict__ eW1z, const float* __restrict__ eb1,
                          const float* __restrict__ eg1, const float* __restrict__ ebe1,
                          const __bf16* __restrict__ eW2z, const float* __restrict__ eb2,
                          const float* __restrict__ eg2, const float* __restrict__ ebe2,
                          float* __restrict__ out_se, int ESn) {
  __shared__ __align__(32) char smem[49152];
  float*  H   = (float*)smem;               // 32 C-frags (32 KB)
  __bf16* Af  = (__bf16*)(smem + 32768);    // 8 A-frags  (8 KB)
  __bf16* Hbf = (__bf16*)(smem + 32768);    // 16 A-frags (reuses Af after layer1)
  float*  O   = (float*)smem;               // 8 C-frags  (8 KB, reuses H)
  int lane = threadIdx.x;
  int e0 = blockIdx.x * 16;

  // stage A = concat(sup[ssrc], sup[sdst]) in fragment order
  for (int idx = lane; idx < 16 * 256; idx += 32) {
    int r = idx >> 8, k = idx & 255;
    int e = e0 + r;
    float v = 0.f;
    if (e < ESn) {
      int node = (k < 128) ? ssrc[e] : sdst[e];
      v = sup[(size_t)node * 128 + (k & 127)];
    }
    Af[(k >> 5) * 512 + a_swz_slot(r, k & 31)] = (__bf16)v;
  }
  __syncthreads();

  const v16bf* AfV = (const v16bf*)Af;
  const v16bf* W1V = (const v16bf*)eW1z;  // KT=8
  // layer 1: H = A[16x256] @ eW1[256x512]  (32 n-tiles x 8 k-steps)
  for (int nt = 0; nt < 32; ++nt) {
    v8f acc = {};
#pragma unroll
    for (int ks = 0; ks < 8; ++ks) {
      v16bf a = AfV[ks * 32 + lane];
      v16bf b = W1V[(size_t)(nt * 8 + ks) * 32 + lane];
      acc = wmma_bf16(a, b, acc);
    }
    store_c_frag(H, nt, lane, acc);
  }
  __syncthreads();

  // bias + LN + ReLU -> bf16 A-fragments
  ln_relu_to_frags_512(H, Hbf, lane, eb1, eg1, ebe1);
  __syncthreads();

  // layer 2: O[16x128] = Hb[16x512] @ eW2[512x128]  (8 n-tiles x 16 k-steps, KT=16)
  const v16bf* HbV = (const v16bf*)Hbf;
  const v16bf* W2V = (const v16bf*)eW2z;
  for (int nt = 0; nt < 8; ++nt) {
    v8f acc = {};
#pragma unroll
    for (int ks = 0; ks < 16; ++ks) {
      v16bf a = HbV[ks * 32 + lane];
      v16bf b = W2V[(size_t)(nt * 16 + ks) * 32 + lane];
      acc = wmma_bf16(a, b, acc);
    }
    store_c_frag(O, nt, lane, acc);
  }
  __syncthreads();

  // final LN over 128 cols: lane owns the 4 contiguous cols c = lane*4+j
  float b2r[4], g2r[4], be2r[4];
#pragma unroll
  for (int j = 0; j < 4; ++j) {
    int c = lane * 4 + j;
    b2r[j] = eb2[c]; g2r[j] = eg2[c]; be2r[j] = ebe2[c];
  }
  for (int r = 0; r < 16; ++r) {
    int e = e0 + r;
    float vv[4];
    float s = 0.f, s2 = 0.f;
#pragma unroll
    for (int j = 0; j < 4; ++j) {
      int c = lane * 4 + j;
      float v = O[c_frag_idx(r, c)] + b2r[j];
      vv[j] = v; s += v; s2 += v * v;
    }
    s = wave_sum(s); s2 = wave_sum(s2);
    float mean = s * (1.f / 128.f);
    float var  = s2 * (1.f / 128.f) - mean * mean;
    float inv  = rsqrtf(var + LN_EPS);
    if (e < ESn) {
      float4 o;
      o.x = fmaxf((vv[0] - mean) * inv * g2r[0] + be2r[0], 0.f);
      o.y = fmaxf((vv[1] - mean) * inv * g2r[1] + be2r[1], 0.f);
      o.z = fmaxf((vv[2] - mean) * inv * g2r[2] + be2r[2], 0.f);
      o.w = fmaxf((vv[3] - mean) * inv * g2r[3] + be2r[3], 0.f);
      *(float4*)(out_se + (size_t)e * 128 + lane * 4) = o;
    }
  }
}

// ---------------------------------------------------------------------------
extern "C" void kernel_launch(void* const* d_in, const int* in_sizes, int n_in,
                              void* d_out, int out_size, void* d_ws, size_t ws_size,
                              hipStream_t stream) {
  const float* nodes   = (const float*)d_in[0];
  const int*   clusters= (const int*)d_in[1];
  const int*   bip_src = (const int*)d_in[2];
  const int*   bip_dst = (const int*)d_in[3];
  const int*   sup_src = (const int*)d_in[4];
  const int*   sup_dst = (const int*)d_in[5];
  const float* cW1 = (const float*)d_in[6];
  const float* cb1 = (const float*)d_in[7];
  const float* cg1 = (const float*)d_in[8];
  const float* cbe1= (const float*)d_in[9];
  const float* cW2 = (const float*)d_in[10];
  const float* cb2 = (const float*)d_in[11];
  const float* eW1 = (const float*)d_in[12];
  const float* eb1 = (const float*)d_in[13];
  const float* eg1 = (const float*)d_in[14];
  const float* ebe1= (const float*)d_in[15];
  const float* eW2 = (const float*)d_in[16];
  const float* eb2 = (const float*)d_in[17];
  const float* eg2 = (const float*)d_in[18];
  const float* ebe2= (const float*)d_in[19];
  (void)n_in; (void)ws_size;

  const int N   = in_sizes[0] / 128;
  const int EBn = in_sizes[2];
  const int ESn = in_sizes[4];
  const int Cn  = (out_size - EBn - ESn * 129) / 128;  // C*128 + ES*128 + EB + ES

  float* out_sup  = (float*)d_out;
  float* out_se   = out_sup + (size_t)Cn * 128;
  float* out_bip  = out_se  + (size_t)ESn * 128;
  float* out_satt = out_bip + (size_t)EBn;

  char* ws = (char*)d_ws;
  size_t off = 0;
  auto alloc = [&](size_t bytes) {
    void* p = ws + off;
    off = (off + bytes + 255) & ~(size_t)255;
    return p;
  };
  float*    emb   = (float*)alloc((size_t)N * 16 * 4);
  float*    att   = (float*)alloc((size_t)EBn * 4);
  float*    sums  = (float*)alloc((size_t)Cn * 16 * 4);
  float*    cnt   = (float*)alloc((size_t)Cn * 4);
  float*    means = (float*)alloc((size_t)Cn * 16 * 4);
  unsigned* amaxe = (unsigned*)alloc((size_t)N * 4);
  unsigned* amine = (unsigned*)alloc((size_t)N * 4);
  float*    asum  = (float*)alloc((size_t)N * 4);
  float*    satt  = (float*)alloc((size_t)ESn * 4);
  unsigned* smaxe = (unsigned*)alloc((size_t)Cn * 4);
  unsigned* smine = (unsigned*)alloc((size_t)Cn * 4);
  __bf16*   cW1z  = (__bf16*)alloc(128 * 512 * 2);
  __bf16*   cW2z  = (__bf16*)alloc(512 * 16 * 2);
  __bf16*   eW1z  = (__bf16*)alloc(256 * 512 * 2);
  __bf16*   eW2z  = (__bf16*)alloc(512 * 128 * 2);

  const int TB = 256;
  init_kernel<<<1024, TB, 0, stream>>>(sums, cnt, asum, out_sup, amaxe, amine, smaxe, smine, Cn, N);
  swz_kernel<<<256, TB, 0, stream>>>(cW1, cW1z, 128, 9);   // [128][512]
  swz_kernel<<<64,  TB, 0, stream>>>(cW2, cW2z, 512, 4);   // [512][16]
  swz_kernel<<<512, TB, 0, stream>>>(eW1, eW1z, 256, 9);   // [256][512]
  swz_kernel<<<256, TB, 0, stream>>>(eW2, eW2z, 512, 7);   // [512][128]

  node_mlp_kernel<<<(N + 15) / 16, 32, 0, stream>>>(nodes, cW1z, cb1, cg1, cbe1, cW2z, cb2, emb, N);

  scatter_sum_kernel<<<(N * 16 + TB - 1) / TB, TB, 0, stream>>>(emb, clusters, sums, cnt, N, Cn);
  means_kernel<<<(Cn + TB - 1) / TB, TB, 0, stream>>>(sums, cnt, means, Cn);

  bip_dot_kernel<<<(EBn + TB - 1) / TB, TB, 0, stream>>>(emb, means, bip_src, bip_dst, att, amaxe, amine, EBn);
  bip_exp_kernel<<<(EBn + TB - 1) / TB, TB, 0, stream>>>(att, amaxe, amine, bip_src, asum, EBn);
  bip_norm_kernel<<<(EBn + TB - 1) / TB, TB, 0, stream>>>(att, asum, bip_src, out_bip, EBn);

  scatter_sup_kernel<<<(int)(((size_t)EBn * 32 + TB - 1) / TB), TB, 0, stream>>>(nodes, att, bip_src, bip_dst, out_sup, EBn);

  sup_dot_kernel<<<(ESn + TB - 1) / TB, TB, 0, stream>>>(means, sup_src, sup_dst, satt, smaxe, smine, ESn);
  sup_tanh_kernel<<<(ESn + TB - 1) / TB, TB, 0, stream>>>(satt, smaxe, smine, sup_dst, out_satt, ESn);

  superedge_mlp_kernel<<<(ESn + 15) / 16, 32, 0, stream>>>(out_sup, sup_src, sup_dst,
                                                           eW1z, eb1, eg1, ebe1,
                                                           eW2z, eb2, eg2, ebe2, out_se, ESn);
}
